// WindowAttention_12180527251719
// MI455X (gfx1250) — matrix operand order repaired and checked by reading, not compiled, not run on hardware
//
#include <hip/hip_runtime.h>
#include <hip/hip_bf16.h>

typedef __attribute__((ext_vector_type(16))) _Float16 v16h;
typedef __attribute__((ext_vector_type(8)))  float    v8f;

#define WH 8
#define WW 8
#define NTOK 64            // window tokens
#define CH 192             // channels
#define NH 6               // heads
#define HD 32              // head dim
#define SCALE 0.17677669529663687f   // 32^-0.5
#define QKVW_ELEMS (CH * 3 * CH)     // 110592
#define PROJW_ELEMS (CH * CH)        // 36864

// ---------------- WMMA fragment helpers (CDNA5 wave32 layouts) ----------------

// A: 16x32 f16, row-major source with given element stride.
// lane: m = L&15, half = L>>4 ; vgpr i holds K = (i>>2)*16 + (i&3)*2 + half*8 (+0,+1)
static __device__ __forceinline__ v16h load_A(const _Float16* base, int stride,
                                              int row0, int k0, int lane) {
  const int m = lane & 15, half = lane >> 4;
  const _Float16* p = base + (row0 + m) * stride + k0 + half * 8;
  v16h a;
#pragma unroll
  for (int i = 0; i < 8; ++i) {
    const int k = ((i >> 2) * 16) + ((i & 3) * 2);
    a[2 * i]     = p[k];
    a[2 * i + 1] = p[k + 1];
  }
  return a;
}

// B: 32x16 f16, row-major source. lane: n = L&15, half = L>>4; element j holds K = (j>>3)*16 + half*8 + (j&7).
static __device__ __forceinline__ v16h load_B(const _Float16* base, int stride,
                                              int k0, int n0, int lane) {
  const int n = lane & 15, half = lane >> 4;
  const _Float16* p = base + k0 * stride + n0 + n;
  v16h b;
#pragma unroll
  for (int j = 0; j < 16; ++j) b[j] = p[((j >> 3) * 16 + half * 8 + (j & 7)) * stride];
  return b;
}

static __device__ __forceinline__ v8f wmma_f16(v16h a, v16h b, v8f c) {
  return __builtin_amdgcn_wmma_f32_16x16x32_f16(false, a, false, b, (short)0, c,
                                                false, false);
}

// ---------------- weight fp32 -> fp16 staging kernel ----------------

__global__ void convert_weights(const float* __restrict__ qkv_w,
                                const float* __restrict__ proj_w,
                                _Float16* __restrict__ qkv_w_h,
                                _Float16* __restrict__ proj_w_h) {
  int i = blockIdx.x * blockDim.x + threadIdx.x;
  if (i < QKVW_ELEMS)  qkv_w_h[i]  = (_Float16)qkv_w[i];
  if (i < PROJW_ELEMS) proj_w_h[i] = (_Float16)proj_w[i];
}

// ---------------- fused window-attention kernel ----------------
// 1 block = 1 window (64 tokens). 256 threads = 8 waves.
// LDS plan (144 KB total):
//   [0)      sXA : 64x192 f16 (input X, later reused as attention output) 24 KB
//   [24576)  Qs  : 6x64x32 f16 (Q * SCALE)                                24 KB
//   [49152)  Kt  : 6x32x64 f16 (K transposed: [d][token])                 24 KB
//   [73728)  Vs  : 6x64x32 f16                                            24 KB
//   [98304)  Sf  : 2x64x64 f32 (scores for one head pair)                 32 KB
//   [131072) Ph  : 2x64x64 f16 (softmax probs for one head pair)          16 KB

__global__ __launch_bounds__(256)
void window_attn_fused(const float* __restrict__ X,        // [8192,64,192]
                       const float* __restrict__ mask,     // [1024,64,64]
                       const float* __restrict__ qkv_b,    // [576]
                       const float* __restrict__ proj_b,   // [192]
                       const float* __restrict__ rel_table,// [225,6]
                       const _Float16* __restrict__ qkv_w_h, // [192,576]
                       const _Float16* __restrict__ proj_w_h,// [192,192]
                       float* __restrict__ out) {          // [8192,64,192]
  __shared__ __align__(16) char smem[147456];
  _Float16* sXA = (_Float16*)(smem);            // 64x192
  _Float16* Qs  = (_Float16*)(smem + 24576);    // [h][64][32]
  _Float16* Kt  = (_Float16*)(smem + 49152);    // [h][32][64]
  _Float16* Vs  = (_Float16*)(smem + 73728);    // [h][64][32]
  float*    Sf  = (float*)   (smem + 98304);    // [2][64][64]
  _Float16* Ph  = (_Float16*)(smem + 131072);   // [2][64][64]

  const int b    = blockIdx.x;
  const int tid  = threadIdx.x;
  // wave id is uniform across the wave: force it into an SGPR so all tile
  // index math and section branches are scalar (s_cbranch, not exec-mask).
  const int wave = __builtin_amdgcn_readfirstlane(tid >> 5);
  const int lane = tid & 31;
  const int fn   = lane & 15;          // fragment N column
  const int fmb  = (lane >> 4) * 8;    // fragment M base

  // hint: pull weight stream toward the WGP early (CDNA5 prefetch path)
  if (tid == 0) {
    __builtin_prefetch(qkv_w_h, 0, 1);
    __builtin_prefetch(proj_w_h, 0, 1);
  }

  // ---- stage 0: load X window, convert to f16 into LDS ----
  {
    const float4* Xg = (const float4*)(X + (size_t)b * (NTOK * CH));
    for (int i = tid; i < (NTOK * CH) / 4; i += 256) {
      float4 v = Xg[i];
      sXA[4 * i + 0] = (_Float16)v.x;
      sXA[4 * i + 1] = (_Float16)v.y;
      sXA[4 * i + 2] = (_Float16)v.z;
      sXA[4 * i + 3] = (_Float16)v.w;
    }
  }
  __syncthreads();

  // ---- stage 1: QKV GEMM  [64,192] x [192,576], scatter into Qs/Kt/Vs ----
  // 4 x 36 = 144 output tiles, K loop = 6 chunks of 32.
  for (int t = wave; t < 144; t += 8) {
    const int mt = __builtin_amdgcn_readfirstlane(t % 4);
    const int nt = __builtin_amdgcn_readfirstlane(t / 4);
    v8f acc = {};
#pragma unroll
    for (int kk = 0; kk < 6; ++kk) {
      v16h a = load_A(sXA, CH, mt * 16, kk * 32, lane);
      v16h w = load_B(qkv_w_h, 3 * CH, kk * 32, nt * 16, lane);
      acc = wmma_f16(a, w, acc);
    }
    const int col0 = nt * 16;
    const int sec  = __builtin_amdgcn_readfirstlane(col0 / CH);  // 0=Q 1=K 2=V
    const int rem  = col0 % CH;
    const int h    = __builtin_amdgcn_readfirstlane(rem / HD);
    const int d    = (rem % HD) + fn;
    const float bias = qkv_b[col0 + fn];
    const int m0 = mt * 16 + fmb;
    if (sec == 0) {
      _Float16* q = Qs + h * (NTOK * HD) + d;
#pragma unroll
      for (int r = 0; r < 8; ++r)
        q[(m0 + r) * HD] = (_Float16)((acc[r] + bias) * SCALE);
    } else if (sec == 1) {
      _Float16* k = Kt + h * (NTOK * HD) + d * NTOK + m0;
#pragma unroll
      for (int r = 0; r < 8; ++r) k[r] = (_Float16)(acc[r] + bias);
    } else {
      _Float16* v = Vs + h * (NTOK * HD) + d;
#pragma unroll
      for (int r = 0; r < 8; ++r)
        v[(m0 + r) * HD] = (_Float16)(acc[r] + bias);
    }
  }
  __syncthreads();

  // ---- stage 2: attention, head pairs ----
  const int w_idx = b & 1023;                       // mask window
  const float* mwin = mask + (size_t)w_idx * (NTOK * NTOK);

  for (int hp = 0; hp < 3; ++hp) {
    // 2a: S = (Q*scale) @ K^T : 2 heads x 4x4 tiles, single K=32 WMMA each
    for (int t = wave; t < 32; t += 8) {
      const int hl = __builtin_amdgcn_readfirstlane(t >> 4);
      const int mt = __builtin_amdgcn_readfirstlane((t >> 2) & 3);
      const int nt = __builtin_amdgcn_readfirstlane(t & 3);
      const int h  = hp * 2 + hl;
      v16h a  = load_A(Qs + h * (NTOK * HD), HD, mt * 16, 0, lane);
      v16h kb = load_B(Kt + h * (NTOK * HD), NTOK, 0, nt * 16, lane);
      v8f acc = {};
      acc = wmma_f16(a, kb, acc);
      float* Sh = Sf + hl * (NTOK * NTOK) + nt * 16 + fn;
#pragma unroll
      for (int r = 0; r < 8; ++r) Sh[(mt * 16 + fmb + r) * NTOK] = acc[r];
    }
    __syncthreads();

    // 2b: + rel-pos bias + mask, row softmax over keys, write packed f16 probs
    if (tid < 128) {
      const int hl = tid >> 6;
      const int n  = tid & 63;
      const int h  = hp * 2 + hl;
      float* row = Sf + hl * (NTOK * NTOK) + n * NTOK;
      const float4* m4 = (const float4*)(mwin + n * NTOK);
      const int ny = n >> 3, nx = n & 7;
      float mx = -3.0e30f;
#pragma unroll
      for (int j = 0; j < 16; ++j) {
        const float4 mv = m4[j];
        const float mvv[4] = {mv.x, mv.y, mv.z, mv.w};
#pragma unroll
        for (int e = 0; e < 4; ++e) {
          const int m = 4 * j + e;
          const int my = m >> 3, mxx = m & 7;
          const int idx = (ny - my + WH - 1) * (2 * WW - 1) + (nx - mxx + WW - 1);
          const float v = row[m] + rel_table[idx * NH + h] + mvv[e];
          row[m] = v;
          mx = fmaxf(mx, v);
        }
      }
      float sum = 0.0f;
#pragma unroll 8
      for (int m = 0; m < NTOK; ++m) {
        const float e = __expf(row[m] - mx);
        row[m] = e;
        sum += e;
      }
      const float inv = 1.0f / sum;
      unsigned int* prow = (unsigned int*)(Ph + hl * (NTOK * NTOK) + n * NTOK);
#pragma unroll 8
      for (int m = 0; m < NTOK; m += 2) {
        union { _Float16 hh[2]; unsigned int u; } pk;
        pk.hh[0] = (_Float16)(row[m] * inv);
        pk.hh[1] = (_Float16)(row[m + 1] * inv);
        prow[m >> 1] = pk.u;
      }
    }
    __syncthreads();

    // 2c: O = P @ V : 2 heads x 4x2 tiles, K loop = 2 chunks of 32
    for (int t = wave; t < 16; t += 8) {
      const int hl = __builtin_amdgcn_readfirstlane(t >> 3);
      const int mt = __builtin_amdgcn_readfirstlane((t >> 1) & 3);
      const int nt = __builtin_amdgcn_readfirstlane(t & 1);
      const int h  = hp * 2 + hl;
      v8f acc = {};
#pragma unroll
      for (int kk = 0; kk < 2; ++kk) {
        v16h a  = load_A(Ph + hl * (NTOK * NTOK), NTOK, mt * 16, kk * 32, lane);
        v16h vb = load_B(Vs + h * (NTOK * HD), HD, kk * 32, nt * 16, lane);
        acc = wmma_f16(a, vb, acc);
      }
      _Float16* o = sXA + h * HD + nt * 16 + fn;
#pragma unroll
      for (int r = 0; r < 8; ++r)
        o[(mt * 16 + fmb + r) * CH] = (_Float16)acc[r];
    }
    __syncthreads();   // protects Sf/Ph reuse next iteration
  }

  // ---- stage 3: output projection  [64,192] x [192,192] + proj_b -> global
  float* outw = out + (size_t)b * (NTOK * CH);
  for (int t = wave; t < 48; t += 8) {
    const int mt = __builtin_amdgcn_readfirstlane(t % 4);
    const int nt = __builtin_amdgcn_readfirstlane(t / 4);
    v8f acc = {};
#pragma unroll
    for (int kk = 0; kk < 6; ++kk) {
      v16h a = load_A(sXA, CH, mt * 16, kk * 32, lane);
      v16h w = load_B(proj_w_h, CH, kk * 32, nt * 16, lane);
      acc = wmma_f16(a, w, acc);
    }
    const float pb = proj_b[nt * 16 + fn];
    float* o = outw + nt * 16 + fn;
#pragma unroll
    for (int r = 0; r < 8; ++r)
      o[(mt * 16 + fmb + r) * CH] = acc[r] + pb;
  }
}

// ---------------- launch ----------------

extern "C" void kernel_launch(void* const* d_in, const int* in_sizes, int n_in,
                              void* d_out, int out_size, void* d_ws, size_t ws_size,
                              hipStream_t stream) {
  const float* hidden = (const float*)d_in[0];
  const float* mask   = (const float*)d_in[1];
  const float* qkv_w  = (const float*)d_in[2];
  const float* qkv_b  = (const float*)d_in[3];
  const float* proj_w = (const float*)d_in[4];
  const float* proj_b = (const float*)d_in[5];
  const float* rtab   = (const float*)d_in[6];
  float* out = (float*)d_out;

  _Float16* qkv_w_h  = (_Float16*)d_ws;                              // 221184 B
  _Float16* proj_w_h = (_Float16*)((char*)d_ws + (size_t)QKVW_ELEMS * 2);

  convert_weights<<<(QKVW_ELEMS + 255) / 256, 256, 0, stream>>>(
      qkv_w, proj_w, qkv_w_h, proj_w_h);

  window_attn_fused<<<8192, 256, 0, stream>>>(
      hidden, mask, qkv_b, proj_b, rtab, qkv_w_h, proj_w_h, out);

  (void)in_sizes; (void)n_in; (void)out_size; (void)ws_size;
}
